// SelfAttention_4303557231303
// MI455X (gfx1250) — matrix operand order repaired and checked by reading, not compiled, not run on hardware
//
#include <hip/hip_runtime.h>
#include <hip/hip_bf16.h>

// ---------------------------------------------------------------------------
// MI455X (gfx1250) feature-space attention, bf16 WMMA + f32 accumulate.
// Memory-bound (~1.4 GB min traffic @ 23.3 TB/s); bf16 intermediates halve it.
// Round 2: k_out keeps its full A-panel in registers (heads read once from
// HBM); k_scores is an async-to-LDS double-buffered tiled GEMM (ASYNCcnt).
// ---------------------------------------------------------------------------

typedef __bf16 bf16_t;
typedef __attribute__((ext_vector_type(16))) __bf16 v16bf;
typedef __attribute__((ext_vector_type(8)))  __bf16 v8bf;
typedef __attribute__((ext_vector_type(8)))  float  v8f;

#define HEADS 8
#define BATCH 32
#define SEQ   4096
#define FEAT  512
#define DIM   64
#define MROWS (BATCH * SEQ)   // 131072 flattened (b,l) rows

union ABfrag { v16bf v; v8bf h[2]; };

// A-fragment (16x32 bf16, M x K) from row-major [M][K] source, tile origin `base`,
// leading dim `ld` (elements). Lane m=lane&15; K chunks {kb..kb+7, kb+16..kb+23}.
__device__ inline v16bf load_a_rm(const bf16_t* base, int ld) {
  const int lane = threadIdx.x & 31;
  const int m  = lane & 15;
  const int kb = (lane >> 4) ? 8 : 0;
  const bf16_t* p = base + (size_t)m * ld + kb;
  ABfrag u;
  u.h[0] = *(const v8bf*)(p);        // K = kb .. kb+7
  u.h[1] = *(const v8bf*)(p + 16);   // K = kb+16 .. kb+23
  return u.v;
}

// B-fragment (32x16 bf16, K x N) from an [N][K] (column-major) source.
// Lane n=lane&15; K chunks {kb..kb+7, kb+16..kb+23}, kb = 8*(lane>=16).
__device__ inline v16bf load_b_cm(const bf16_t* base, int ld) {
  const int lane = threadIdx.x & 31;
  const int n  = lane & 15;
  const int kb = (lane >> 4) ? 8 : 0;
  const bf16_t* p = base + (size_t)n * ld + kb;
  ABfrag u;
  u.h[0] = *(const v8bf*)(p);
  u.h[1] = *(const v8bf*)(p + 16);
  return u.v;
}

__device__ inline v8f wmma_bf16(v16bf a, v16bf b, v8f c) {
  return __builtin_amdgcn_wmma_f32_16x16x32_bf16(false, a, false, b,
                                                 (short)0, c, false, false);
}

// Issue one per-lane async global->LDS 128-bit copy (tracked by ASYNCcnt).
__device__ inline void async_copy_b128(const bf16_t* gsrc, bf16_t* ldst) {
  unsigned long long ga = (unsigned long long)(uintptr_t)gsrc;
  unsigned int la = (unsigned int)(uintptr_t)ldst;   // generic addr[31:0] == LDS offset
  asm volatile("global_load_async_to_lds_b128 %0, %1, off"
               :: "v"(la), "v"(ga) : "memory");
}
__device__ inline void wait_async0() {
  asm volatile("s_wait_asynccnt 0" ::: "memory");
}

// --------------------------- kernel 0: weight prep -------------------------
// src f32 [batch][R][C] -> dst bf16 [batch][C][R]  (i.e. [N][K] for WMMA B)
__global__ void k_wt(const float* __restrict__ src, bf16_t* __restrict__ dst,
                     int R, int C, int total) {
  int i = blockIdx.x * blockDim.x + threadIdx.x;
  if (i >= total) return;
  int c  = i % C;
  int r  = (i / C) % R;
  int bt = i / (C * R);
  dst[(size_t)bt * R * C + (size_t)c * R + r] = (bf16_t)src[i];
}

// --------------------------- kernel 1: QKV projections ---------------------
// One block = 32 rows of x staged in LDS (bf16), reused by 3 matrices x 8 heads.
// Q,K stored transposed [h][b][d][l]; V stored natural [h][b][l][d].
__global__ void __launch_bounds__(256) k_qkv(
    const float* __restrict__ x,
    const bf16_t* __restrict__ WqT, const bf16_t* __restrict__ WkT,
    const bf16_t* __restrict__ WvT,
    const float* __restrict__ bq, const float* __restrict__ bk,
    const float* __restrict__ bv,
    bf16_t* __restrict__ Qt, bf16_t* __restrict__ Kt, bf16_t* __restrict__ V)
{
  __shared__ __align__(16) bf16_t xs[32][FEAT + 8];
  const int m0 = blockIdx.x * 32;          // flattened (b,l) row base

  for (int i = threadIdx.x; i < 32 * (FEAT / 4); i += 256) {
    int row = i >> 7;                      // FEAT/4 = 128 float4 per row
    int c4  = i & 127;
    float4 f = ((const float4*)(x + (size_t)(m0 + row) * FEAT))[c4];
    bf16_t* d = &xs[row][c4 * 4];
    d[0] = (bf16_t)f.x; d[1] = (bf16_t)f.y;
    d[2] = (bf16_t)f.z; d[3] = (bf16_t)f.w;
  }
  __syncthreads();

  const int wave = threadIdx.x >> 5;
  const int lane = threadIdx.x & 31;
  const int b  = m0 >> 12;                 // SEQ = 4096; block stays in one b
  const int l0 = m0 & (SEQ - 1);

  for (int t = wave; t < 192; t += 8) {    // 3 mats * 8 heads * 2 mtiles * 4 ntiles
    const int nt  = t & 3;
    const int mt  = (t >> 2) & 1;
    const int h   = (t >> 3) & 7;
    const int mat = t >> 6;
    const bf16_t* WT  = (mat == 0) ? WqT : (mat == 1) ? WkT : WvT;
    const float* bias = (mat == 0) ? bq  : (mat == 1) ? bk  : bv;

    v8f c = {0.f,0.f,0.f,0.f,0.f,0.f,0.f,0.f};
    const bf16_t* wbase = WT + ((size_t)h * DIM + nt * 16) * FEAT;
    const int am = mt * 16 + (lane & 15);
    const int kb = (lane >> 4) ? 8 : 0;
    for (int kk = 0; kk < FEAT; kk += 32) {
      const bf16_t* ap = &xs[am][kk + kb];
      ABfrag ua; ua.h[0] = *(const v8bf*)ap; ua.h[1] = *(const v8bf*)(ap + 16);
      v16bf bf = load_b_cm(wbase + kk, FEAT);
      c = wmma_bf16(ua.v, bf, c);
    }
    const float bb = bias[h * DIM + nt * 16 + (lane & 15)];
    #pragma unroll
    for (int r = 0; r < 8; ++r) c[r] += bb;

    const int dloc = nt * 16 + (lane & 15);
    const int mh   = (lane >> 4) ? 8 : 0;
    if (mat < 2) {                         // transposed bf16 store, 8 contiguous l
      bf16_t* dstT = (mat == 0 ? Qt : Kt)
        + (((size_t)h * BATCH + b) * DIM + dloc) * SEQ + (l0 + mt * 16 + mh);
      v8bf pk;
      #pragma unroll
      for (int r = 0; r < 8; ++r) pk[r] = (bf16_t)c[r];
      *(v8bf*)dstT = pk;
    } else {                               // V natural [l][d]
      bf16_t* dstV = V + ((size_t)h * BATCH + b) * SEQ * DIM;
      #pragma unroll
      for (int r = 0; r < 8; ++r)
        dstV[(size_t)(l0 + mt * 16 + mh + r) * DIM + dloc] = (bf16_t)c[r];
    }
  }
}

// --------------------------- kernel 2: scores = Q^T K / 8 ------------------
// One block per (h,b); contraction over L=4096. Async-to-LDS double-buffered:
// per iteration, 256 threads issue async B128 copies of the NEXT 64x32 Q/K
// panels while waves run WMMA out of the current LDS buffers. Qt/Kt are read
// exactly once per block. Writes S^T f32 [e][d].
__global__ void __launch_bounds__(256) k_scores(
    const bf16_t* __restrict__ Qt, const bf16_t* __restrict__ Kt,
    float* __restrict__ St)
{
  __shared__ __align__(16) bf16_t Qs[2][64][40];   // 32 K + 8 pad
  __shared__ __align__(16) bf16_t Ks[2][64][40];
  const int hb   = blockIdx.x;
  const int tid  = threadIdx.x;
  const int wave = tid >> 5;
  const int lane = tid & 31;
  const bf16_t* Qb = Qt + (size_t)hb * DIM * SEQ;
  const bf16_t* Kb = Kt + (size_t)hb * DIM * SEQ;

  // Wave w owns tiles (mt, nt0) and (mt, nt0+2): same A-fragment, two B's.
  const int mt   = wave & 3;
  const int nt0  = wave >> 2;
  const int arow = mt * 16 + (lane & 15);
  const int akb  = (lane >> 4) ? 8 : 0;
  const int brow = nt0 * 16 + (lane & 15);
  const int bkb  = (lane >> 4) ? 8 : 0;

  // Stage panels Q[0:64][kk:kk+32], K[0:64][kk:kk+32] into buffer `buf`.
  // 128 rows * 64 B = 512 b128 chunks -> 2 per thread.
  auto stage = [&](int buf, int kk) {
    #pragma unroll
    for (int i = 0; i < 2; ++i) {
      int c   = tid + i * 256;
      int m   = c >> 8;            // 0 = Q, 1 = K
      int row = (c >> 2) & 63;
      int seg = c & 3;             // 8-element (16B) segment within 32 K
      const bf16_t* g = (m ? Kb : Qb) + (size_t)row * SEQ + kk + seg * 8;
      bf16_t* l = m ? &Ks[buf][row][seg * 8] : &Qs[buf][row][seg * 8];
      async_copy_b128(g, l);
    }
  };

  v8f c0 = {0.f,0.f,0.f,0.f,0.f,0.f,0.f,0.f};
  v8f c1 = c0;

  stage(0, 0);
  const int NIT = SEQ / 32;        // 128
  for (int it = 0; it < NIT; ++it) {
    const int buf = it & 1;
    wait_async0();                 // my async stores into `buf` are done
    __syncthreads();               // everyone's are; prev buffer fully consumed
    if (it + 1 < NIT) stage(buf ^ 1, (it + 1) * 32);

    ABfrag ua;
    const bf16_t* ap = &Qs[buf][arow][akb];
    ua.h[0] = *(const v8bf*)ap; ua.h[1] = *(const v8bf*)(ap + 16);
    ABfrag ub0;
    const bf16_t* bp0 = &Ks[buf][brow][bkb];
    ub0.h[0] = *(const v8bf*)bp0; ub0.h[1] = *(const v8bf*)(bp0 + 16);
    ABfrag ub1;
    const bf16_t* bp1 = &Ks[buf][brow + 32][bkb];
    ub1.h[0] = *(const v8bf*)bp1; ub1.h[1] = *(const v8bf*)(bp1 + 16);
    c0 = wmma_bf16(ua.v, ub0.v, c0);
    c1 = wmma_bf16(ua.v, ub1.v, c1);
  }

  #pragma unroll
  for (int r = 0; r < 8; ++r) { c0[r] *= 0.125f; c1[r] *= 0.125f; }  // 1/sqrt(64)
  const int e0 = nt0 * 16 + (lane & 15);
  const int e1 = (nt0 + 2) * 16 + (lane & 15);
  const int d0 = mt * 16 + ((lane >> 4) ? 8 : 0);
  *(v8f*)(St + ((size_t)hb * DIM + e0) * DIM + d0) = c0;  // S^T: row e, contig d
  *(v8f*)(St + ((size_t)hb * DIM + e1) * DIM + d0) = c1;
}

// --------------------------- kernel 3: softmax over d ----------------------
// S^T rows are contiguous over d. One wave per 64-element row. Writes bf16 A^T.
__global__ void __launch_bounds__(256) k_softmax(
    const float* __restrict__ St, bf16_t* __restrict__ At)
{
  const int wave = threadIdx.x >> 5;
  const int lane = threadIdx.x & 31;
  const int row  = blockIdx.x * 8 + wave;          // HB*64 = 16384 rows
  const float* p = St + (size_t)row * DIM;
  float v0 = p[lane], v1 = p[lane + 32];
  float m = fmaxf(v0, v1);
  #pragma unroll
  for (int off = 16; off >= 1; off >>= 1) m = fmaxf(m, __shfl_xor(m, off, 32));
  float e0 = __expf(v0 - m), e1 = __expf(v1 - m);
  float s = e0 + e1;
  #pragma unroll
  for (int off = 16; off >= 1; off >>= 1) s += __shfl_xor(s, off, 32);
  float inv = 1.0f / s;
  At[(size_t)row * DIM + lane]      = (bf16_t)(e0 * inv);
  At[(size_t)row * DIM + lane + 32] = (bf16_t)(e1 * inv);
}

// --------------------------- kernel 4: heads = V @ A -----------------------
// A^T[e][d] is already the column-major B operand. heads stored flat [H][B][L][D].
__global__ void __launch_bounds__(256) k_heads(
    const bf16_t* __restrict__ V, const bf16_t* __restrict__ At,
    bf16_t* __restrict__ Hd)
{
  const int hb   = blockIdx.x >> 5;
  const int lblk = (blockIdx.x & 31) * 128;
  const int wave = threadIdx.x >> 5;
  const int lane = threadIdx.x & 31;
  const int m0   = lblk + wave * 16;
  const bf16_t* Vb = V  + (size_t)hb * SEQ * DIM + (size_t)m0 * DIM;
  const bf16_t* Ab = At + (size_t)hb * DIM * DIM;
  bf16_t*       Hb = Hd + (size_t)hb * SEQ * DIM;

  v8f z = {0.f,0.f,0.f,0.f,0.f,0.f,0.f,0.f};
  v8f c[4] = {z, z, z, z};
  for (int kk = 0; kk < DIM; kk += 32) {
    v16bf a = load_a_rm(Vb + kk, DIM);
    #pragma unroll
    for (int nt = 0; nt < 4; ++nt) {
      v16bf b = load_b_cm(Ab + (size_t)(nt * 16) * DIM + kk, DIM);
      c[nt] = wmma_bf16(a, b, c[nt]);
    }
  }
  const int mh = (lane >> 4) ? 8 : 0;
  #pragma unroll
  for (int nt = 0; nt < 4; ++nt) {
    const int e = nt * 16 + (lane & 15);
    #pragma unroll
    for (int r = 0; r < 8; ++r)
      Hb[(size_t)(m0 + mh + r) * DIM + e] = (bf16_t)c[nt][r];
  }
}

// --------------------------- kernel 5: out = heads_flat @ Wo + bo ----------
// Pure reinterpretation: [131072x512] x [512x512]. Each wave loads its entire
// K=512 A-panel into registers ONCE (16 frags = 128 VGPRs), then sweeps all
// 8 n-blocks: heads is read exactly once from HBM; WoT stays L2-resident.
__global__ void __launch_bounds__(256) k_out(
    const bf16_t* __restrict__ Hd, const bf16_t* __restrict__ WoT,
    const float* __restrict__ bo, float* __restrict__ out)
{
  const int m0 = blockIdx.x * 128 + (threadIdx.x >> 5) * 16;
  const int lane = threadIdx.x & 31;
  const bf16_t* Ab = Hd + (size_t)m0 * FEAT;

  v16bf a[16];
  #pragma unroll
  for (int k16 = 0; k16 < 16; ++k16) a[k16] = load_a_rm(Ab + k16 * 32, FEAT);

  const int mh = (lane >> 4) ? 8 : 0;
  v8f z = {0.f,0.f,0.f,0.f,0.f,0.f,0.f,0.f};
  for (int nb = 0; nb < 8; ++nb) {
    const int n0 = nb * 64;
    v8f c[4] = {z, z, z, z};
    #pragma unroll
    for (int k16 = 0; k16 < 16; ++k16) {
      #pragma unroll
      for (int nt = 0; nt < 4; ++nt) {
        v16bf b = load_b_cm(WoT + (size_t)(n0 + nt * 16) * FEAT + k16 * 32, FEAT);
        c[nt] = wmma_bf16(a[k16], b, c[nt]);
      }
    }
    #pragma unroll
    for (int nt = 0; nt < 4; ++nt) {
      const int col  = n0 + nt * 16 + (lane & 15);
      const float bb = bo[col];
      #pragma unroll
      for (int r = 0; r < 8; ++r)
        out[(size_t)(m0 + mh + r) * FEAT + col] = c[nt][r] + bb;
    }
  }
}

// ---------------------------------------------------------------------------
extern "C" void kernel_launch(void* const* d_in, const int* in_sizes, int n_in,
                              void* d_out, int out_size, void* d_ws, size_t ws_size,
                              hipStream_t stream) {
  const float* x  = (const float*)d_in[0];
  const float* Wq = (const float*)d_in[1];
  const float* bq = (const float*)d_in[2];
  const float* Wk = (const float*)d_in[3];
  const float* bk = (const float*)d_in[4];
  const float* Wv = (const float*)d_in[5];
  const float* bv = (const float*)d_in[6];
  const float* Wo = (const float*)d_in[7];
  const float* bo = (const float*)d_in[8];
  float* out = (float*)d_out;

  char* ws = (char*)d_ws;
  size_t off = 0;
  auto carve = [&](size_t bytes) -> void* {
    void* p = ws + off;
    off = (off + bytes + 255) & ~(size_t)255;
    return p;
  };
  const size_t szW   = (size_t)HEADS * DIM * FEAT * sizeof(bf16_t);   // 512 KB
  const size_t szWo  = (size_t)FEAT * FEAT * sizeof(bf16_t);          // 512 KB
  const size_t szQKV = (size_t)HEADS * BATCH * SEQ * DIM * sizeof(bf16_t); // 128 MB
  const size_t szSt  = (size_t)HEADS * BATCH * DIM * DIM * sizeof(float);
  const size_t szAt  = (size_t)HEADS * BATCH * DIM * DIM * sizeof(bf16_t);

  bf16_t* WqT = (bf16_t*)carve(szW);
  bf16_t* WkT = (bf16_t*)carve(szW);
  bf16_t* WvT = (bf16_t*)carve(szW);
  bf16_t* WoT = (bf16_t*)carve(szWo);
  bf16_t* Qt  = (bf16_t*)carve(szQKV);
  bf16_t* Kt  = (bf16_t*)carve(szQKV);
  bf16_t* V   = (bf16_t*)carve(szQKV);
  float*  St  = (float*) carve(szSt);
  bf16_t* At  = (bf16_t*)carve(szAt);
  bf16_t* Hd  = (bf16_t*)carve(szQKV);
  (void)ws_size; (void)n_in; (void)in_sizes; (void)out_size;

  // 0) weight transpose + f32->bf16 (tiny)
  {
    int total = HEADS * FEAT * DIM;                       // 262144
    k_wt<<<(total + 255) / 256, 256, 0, stream>>>(Wq, WqT, FEAT, DIM, total);
    k_wt<<<(total + 255) / 256, 256, 0, stream>>>(Wk, WkT, FEAT, DIM, total);
    k_wt<<<(total + 255) / 256, 256, 0, stream>>>(Wv, WvT, FEAT, DIM, total);
    int totalo = FEAT * FEAT;                             // 262144
    k_wt<<<(totalo + 255) / 256, 256, 0, stream>>>(Wo, WoT, FEAT, FEAT, totalo);
  }
  // 1) QKV projections (x read from HBM exactly once, staged in LDS)
  k_qkv<<<MROWS / 32, 256, 0, stream>>>(x, WqT, WkT, WvT, bq, bk, bv, Qt, Kt, V);
  // 2) scores S^T = (Q^T K)/8, async-LDS double-buffered, per (h,b)
  k_scores<<<HEADS * BATCH, 256, 0, stream>>>(Qt, Kt, St);
  // 3) softmax over d (contiguous rows of S^T), output bf16 A^T
  k_softmax<<<(HEADS * BATCH * DIM) / 8, 256, 0, stream>>>(St, At);
  // 4) heads = V @ A, flat [H][B][L][D]
  k_heads<<<HEADS * BATCH * (SEQ / 128), 256, 0, stream>>>(V, At, Hd);
  // 5) out = heads_flat @ Wo + bo (heads read once; A-panel register-resident)
  k_out<<<MROWS / 128, 256, 0, stream>>>(Hd, WoT, bo, out);
}
